// Encoder_412316860477
// MI455X (gfx1250) — compile-verified, matrix operand order checked by reading, not run on hardware
//
#include <hip/hip_runtime.h>
#include <hip/hip_bf16.h>
#include <math.h>

// ---------------------------------------------------------------------------
// Problem constants (from reference)
// ---------------------------------------------------------------------------
#define B_SZ    8192
#define P_SZ    128
#define HID     128
#define NF      4
#define LAT     16
#define NPIECES 20

// Workspace layout (units: _Float16 elements unless noted)
#define W1T_OFF 0        // f16 [128][32]   (K zero-padded 4 -> 32), [N][K]
#define W2T_OFF 4096     // f16 [128][128]  [N][K]
#define W3T_OFF 20480    // f16 [32][128]   [N][K]
#define WBYTES  49152    // total f16 weight pack bytes
#define WPOOL_BYTE_OFF 49152  // f32 [32][128] FSPool weights (in ws)

// LDS layout (bytes)
#define LDS_W    0                 // 49152 B: weight pack (same layout as ws)
#define LDS_X    49152             // 2048 B: x row staged as f32[128][4]
#define LDS_H1   51200             // 32768 B: h1 f16 [128][128]
#define LDS_H2   83968             // 32768 B: h2 f16 [128][128]
#define LDS_Z    51200             // 16384 B: z f32 [32][128] (aliases dead h1)
#define LDS_POOL 67584             // 128 B: pooled[32] (inside dead h1)
#define LDS_TOTAL 116736

typedef __attribute__((ext_vector_type(16))) _Float16 v16h;
typedef __attribute__((ext_vector_type(8)))  _Float16 v8h;
typedef __attribute__((ext_vector_type(8)))  float    v8f;

union F16Frag { v16h v; v8h h[2]; };

// Load a 16x32 f16 A/B fragment slice for this lane.
// rowbase_k points at &buf[row * stride + ktile*32]; per ISA layout the lane
// holds K = {kh*8 .. kh*8+7} in elements 0..7 and K = {16+kh*8 ..} in 8..15.
__device__ __forceinline__ v16h ld_frag16(const _Float16* rowbase_k, int kh) {
  F16Frag u;
  u.h[0] = *(const v8h*)(rowbase_k + kh * 8);
  u.h[1] = *(const v8h*)(rowbase_k + kh * 8 + 16);
  return u.v;
}

// ---- CDNA5 async global->LDS copy (ASYNCcnt-tracked), 16 B per lane --------
// ISA 15.18.3: GLOBAL_LOAD_ASYNC_TO_LDS_B128  (GV mode: 64-bit VGPR address)
//   vdst = VGPR holding LDS byte address, vaddr = VGPR pair with global addr.
__device__ __forceinline__ void async_copy16(const void* g, void* l) {
  unsigned loff = (unsigned)(unsigned long long)
      (__attribute__((address_space(3))) char*)l;
  asm volatile("global_load_async_to_lds_b128 %0, %1, off"
               :: "v"(loff), "v"(g) : "memory");
}

__device__ __forceinline__ void wait_async0() {
#if __has_builtin(__builtin_amdgcn_s_wait_asynccnt)
  __builtin_amdgcn_s_wait_asynccnt(0);
#else
  asm volatile("s_wait_asynccnt 0x0" ::: "memory");
#endif
}

// ---------------------------------------------------------------------------
// Prep kernel: convert/transposed-pack weights to f16 in ws, build FSPool table
// ---------------------------------------------------------------------------
__global__ void prep_kernel(const float* __restrict__ W1,
                            const float* __restrict__ W2,
                            const float* __restrict__ W3,
                            const float* __restrict__ pw,
                            _Float16* __restrict__ ws16,
                            float* __restrict__ wpool) {
  int i = blockIdx.x * blockDim.x + threadIdx.x;
  if (i < 4096) {                       // W1 -> [n][k], k padded to 32
    int n = i >> 5, k = i & 31;
    ws16[W1T_OFF + i] = (k < NF) ? (_Float16)W1[k * HID + n] : (_Float16)0.0f;
  } else if (i < 20480) {               // W2 -> [n][k]
    int j = i - 4096; int n = j >> 7, k = j & 127;
    ws16[W2T_OFF + j] = (_Float16)W2[k * HID + n];
  } else if (i < 24576) {               // W3 -> [n][k], n in [0,32)
    int j = i - 20480; int n = j >> 7, k = j & 127;
    ws16[W3T_OFF + j] = (_Float16)W3[k * (2 * LAT) + n];
  } else if (i < 28672) {               // FSPool piecewise-linear weights
    int j = i - 24576; int c = j >> 7, p = j & 127;
    float pos = (float)p / (float)(P_SZ - 1) * (float)NPIECES;
    int idx = (int)pos;
    if (idx < 0) idx = 0;
    if (idx > NPIECES) idx = NPIECES;
    float frac = pos - (float)idx;
    int idx2 = idx + 1; if (idx2 > NPIECES) idx2 = NPIECES;
    wpool[j] = (1.0f - frac) * pw[c * (NPIECES + 1) + idx]
             +          frac * pw[c * (NPIECES + 1) + idx2];
  }
}

// ---------------------------------------------------------------------------
// Fused encoder: async weight stage -> MLP (WMMA) -> transpose -> bitonic sort
// -> FSPool -> reparameterize.  One workgroup (8 wave32) per batch row.
// ---------------------------------------------------------------------------
__global__ void __launch_bounds__(256)
encoder_kernel(const float* __restrict__ x,
               const float* __restrict__ b1,
               const float* __restrict__ b2,
               const float* __restrict__ b3,
               const float* __restrict__ eps,
               const _Float16* __restrict__ ws16,
               const float* __restrict__ wpool,
               float* __restrict__ out) {
  __shared__ __align__(16) unsigned char smem[LDS_TOTAL];
  _Float16* Wlds = (_Float16*)(smem + LDS_W);
  float*    xs   = (float*)(smem + LDS_X);
  _Float16* h1   = (_Float16*)(smem + LDS_H1);
  _Float16* h2   = (_Float16*)(smem + LDS_H2);
  float*    zbuf = (float*)(smem + LDS_Z);
  float*    pooled = (float*)(smem + LDS_POOL);

  const int tid  = threadIdx.x;
  const int wave = tid >> 5;    // M-tile index (16 points each)
  const int lane = tid & 31;
  const int nl   = lane & 15;   // row (A) / col (B,C) local index
  const int kh   = lane >> 4;   // which K-half of the fragment
  const int b    = blockIdx.x;

  // ---- Stage weight pack (48 KB) + this row's x (2 KB) into LDS, async ----
  {
    const char* gw = (const char*)ws16;
    for (int off = tid * 16; off < WBYTES; off += 256 * 16)
      async_copy16(gw + off, (char*)smem + LDS_W + off);
    const char* gx = (const char*)(x + (size_t)b * (P_SZ * NF));
    if (tid < 128)
      async_copy16(gx + tid * 16, (char*)smem + LDS_X + tid * 16);
  }
  wait_async0();
  __syncthreads();

  const _Float16* W1t = Wlds + W1T_OFF;
  const _Float16* W2t = Wlds + W2T_OFF;
  const _Float16* W3t = Wlds + W3T_OFF;

  // ---- Layer 1: x(16x4, K-padded to 32) @ W1 -> h1 strip (16x128) ----
  v16h a1 = {};
  if (kh == 0) {   // only K=0..3 are real features
    const float* xp = xs + (wave * 16 + nl) * NF;
    a1[0] = (_Float16)xp[0];
    a1[1] = (_Float16)xp[1];
    a1[2] = (_Float16)xp[2];
    a1[3] = (_Float16)xp[3];
  }
#pragma unroll
  for (int nt = 0; nt < 8; ++nt) {
    v16h bf = ld_frag16(W1t + (nt * 16 + nl) * 32, kh);
    v8f c = {};
    c = __builtin_amdgcn_wmma_f32_16x16x32_f16(false, a1, false, bf,
                                               (short)0, c, false, false);
    float bias = b1[nt * 16 + nl];
#pragma unroll
    for (int r = 0; r < 8; ++r) {
      float v = c[r] + bias;
      v = v > 0.0f ? v : 0.0f;                                   // ReLU
      h1[(wave * 16 + r + kh * 8) * HID + nt * 16 + nl] = (_Float16)v;
    }
  }
  __syncthreads();

  // ---- Layer 2: h1(16x128) @ W2(128x128) -> h2 strip ----
  v16h a2[4];
#pragma unroll
  for (int kt = 0; kt < 4; ++kt)
    a2[kt] = ld_frag16(h1 + (wave * 16 + nl) * HID + kt * 32, kh);
#pragma unroll
  for (int nt = 0; nt < 8; ++nt) {
    v8f c = {};
#pragma unroll
    for (int kt = 0; kt < 4; ++kt) {
      v16h bf = ld_frag16(W2t + (nt * 16 + nl) * HID + kt * 32, kh);
      c = __builtin_amdgcn_wmma_f32_16x16x32_f16(false, a2[kt], false, bf,
                                                 (short)0, c, false, false);
    }
    float bias = b2[nt * 16 + nl];
#pragma unroll
    for (int r = 0; r < 8; ++r) {
      float v = c[r] + bias;
      v = v > 0.0f ? v : 0.0f;                                   // ReLU
      h2[(wave * 16 + r + kh * 8) * HID + nt * 16 + nl] = (_Float16)v;
    }
  }
  __syncthreads();

  // ---- Layer 3: h2(16x128) @ W3(128x32) -> z transposed [channel][point] ----
  v16h a3[4];
#pragma unroll
  for (int kt = 0; kt < 4; ++kt)
    a3[kt] = ld_frag16(h2 + (wave * 16 + nl) * HID + kt * 32, kh);
#pragma unroll
  for (int nt = 0; nt < 2; ++nt) {
    v8f c = {};
#pragma unroll
    for (int kt = 0; kt < 4; ++kt) {
      v16h bf = ld_frag16(W3t + (nt * 16 + nl) * HID + kt * 32, kh);
      c = __builtin_amdgcn_wmma_f32_16x16x32_f16(false, a3[kt], false, bf,
                                                 (short)0, c, false, false);
    }
    int ch = nt * 16 + nl;
    float bias = b3[ch];
#pragma unroll
    for (int r = 0; r < 8; ++r)
      zbuf[ch * P_SZ + wave * 16 + r + kh * 8] = c[r] + bias;    // transpose store
  }
  __syncthreads();

  // ---- Bitonic sort (descending) of each channel's 128 values ----
  // 28 substages; 32 ch x 64 pairs = 2048 compare-exchanges -> 8 per thread.
  for (int k = 2; k <= P_SZ; k <<= 1) {
    for (int j = k >> 1; j >= 1; j >>= 1) {
      for (int t = tid; t < 32 * 64; t += 256) {
        int ch = t >> 6, q = t & 63;
        int i0 = ((q & ~(j - 1)) << 1) | (q & (j - 1));  // bit j clear
        int i1 = i0 | j;
        float va = zbuf[ch * P_SZ + i0];
        float vb = zbuf[ch * P_SZ + i1];
        bool desc = ((i0 & k) == 0);
        bool sw = desc ? (va < vb) : (va > vb);
        if (sw) {
          zbuf[ch * P_SZ + i0] = vb;
          zbuf[ch * P_SZ + i1] = va;
        }
      }
      __syncthreads();
    }
  }

  // ---- FSPool dot products ----
  if (tid < 32) {
    const float* zp = zbuf + tid * P_SZ;
    const float* wp = wpool + tid * P_SZ;
    float s = 0.0f;
    for (int p = 0; p < P_SZ; ++p) s += zp[p] * wp[p];
    pooled[tid] = s;
  }
  __syncthreads();

  // ---- Reparameterize and write mus / logvars / samples ----
  if (tid < LAT) {
    float mu = pooled[2 * tid];
    float lv = pooled[2 * tid + 1];
    float sm = mu + eps[(size_t)b * LAT + tid] * expf(0.5f * lv);
    out[(size_t)b * LAT + tid] = mu;
    out[(size_t)B_SZ * LAT + (size_t)b * LAT + tid] = lv;
    out[2 * (size_t)B_SZ * LAT + (size_t)b * LAT + tid] = sm;
  }
}

// ---------------------------------------------------------------------------
// Launch
// ---------------------------------------------------------------------------
extern "C" void kernel_launch(void* const* d_in, const int* in_sizes, int n_in,
                              void* d_out, int out_size, void* d_ws, size_t ws_size,
                              hipStream_t stream) {
  const float* x   = (const float*)d_in[0];
  const float* W1  = (const float*)d_in[1];
  const float* b1  = (const float*)d_in[2];
  const float* W2  = (const float*)d_in[3];
  const float* b2  = (const float*)d_in[4];
  const float* W3  = (const float*)d_in[5];
  const float* b3  = (const float*)d_in[6];
  const float* pw  = (const float*)d_in[7];
  const float* eps = (const float*)d_in[8];

  _Float16* ws16 = (_Float16*)d_ws;
  float* wpool   = (float*)((char*)d_ws + WPOOL_BYTE_OFF);
  float* out     = (float*)d_out;

  // 28672 prep elements -> 112 blocks of 256 threads
  prep_kernel<<<112, 256, 0, stream>>>(W1, W2, W3, pw, ws16, wpool);
  encoder_kernel<<<B_SZ, 256, 0, stream>>>(x, b1, b2, b3, eps, ws16, wpool, out);
}
